// TITANsEncoder_53042846106016
// MI455X (gfx1250) — matrix operand order repaired and checked by reading, not compiled, not run on hardware
//
#include <hip/hip_runtime.h>
#include <hip/hip_bf16.h>

// ---------------------------------------------------------------------------
// TITANs encoder forward, CDNA5 (gfx1250) implementation.
// Mixed precision: bf16 WMMA GEMMs (v_wmma_f32_16x16x32_bf16) with fp32
// accumulate; fp32 residual stream / layernorm statistics.
// ---------------------------------------------------------------------------

#define BB 2
#define SS 2048
#define DD 512
#define LL 6
#define HH 8
#define WW 64
#define MHH 4
#define CHH 64
#define NPTT 16

typedef __bf16 bf16;
typedef __attribute__((ext_vector_type(16))) __bf16 v16bf;
typedef __attribute__((ext_vector_type(8)))  __bf16 v8bf;
typedef __attribute__((ext_vector_type(8)))  float  v8f;

__device__ __forceinline__ v16bf mk16(v8bf lo, v8bf hi) {
    union { v16bf v; v8bf h[2]; } u;
    u.h[0] = lo; u.h[1] = hi;
    return u.v;
}

__device__ __forceinline__ v8f wmma_bf16(v16bf a, v16bf b, v8f c) {
    // D = A(16x32 bf16) * B(32x16 bf16) + C(16x16 f32)
    return __builtin_amdgcn_wmma_f32_16x16x32_bf16(
        false, a, false, b, (short)0, c, false, false);
}

__device__ __forceinline__ float gelu_tanh(float x) {
    float x3 = x * x * x;
    float t = tanhf(0.7978845608028654f * (x + 0.044715f * x3));
    return 0.5f * x * (1.0f + t);
}

// ---------------------------------------------------------------------------
// Weight transpose + fp32 -> bf16:  WT[n*K + k] = W[k*N + n]
// ---------------------------------------------------------------------------
__global__ void wtrans_k(const float* __restrict__ W, bf16* __restrict__ WT,
                         int K, int N) {
    size_t i = (size_t)blockIdx.x * blockDim.x + threadIdx.x;
    size_t total = (size_t)K * N;
    if (i >= total) return;
    size_t n = i / K, k = i % K;
    WT[n * (size_t)K + k] = (bf16)W[k * (size_t)N + n];
}

// ---------------------------------------------------------------------------
// LayerNorm over D=512, one 128-thread block per row.
// Writes fp32 (optional) and bf16 outputs.
// ---------------------------------------------------------------------------
__global__ void layernorm_k(const float* __restrict__ x, const float* __restrict__ w,
                            const float* __restrict__ b, float* __restrict__ outf,
                            bf16* __restrict__ outb) {
    __shared__ float red[128];
    int row = blockIdx.x;
    int t = threadIdx.x;
    const float* xr = x + (size_t)row * DD;
    float v0[4];
    float s = 0.0f;
#pragma unroll
    for (int i = 0; i < 4; i++) { v0[i] = xr[t + i * 128]; s += v0[i]; }
    red[t] = s; __syncthreads();
    for (int o = 64; o > 0; o >>= 1) { if (t < o) red[t] += red[t + o]; __syncthreads(); }
    float mean = red[0] * (1.0f / DD);
    __syncthreads();
    float q = 0.0f;
#pragma unroll
    for (int i = 0; i < 4; i++) { float d = v0[i] - mean; q += d * d; }
    red[t] = q; __syncthreads();
    for (int o = 64; o > 0; o >>= 1) { if (t < o) red[t] += red[t + o]; __syncthreads(); }
    float rstd = rsqrtf(red[0] * (1.0f / DD) + 1e-5f);
#pragma unroll
    for (int i = 0; i < 4; i++) {
        int c = t + i * 128;
        float y = (v0[i] - mean) * rstd * w[c] + b[c];
        if (outf) outf[(size_t)row * DD + c] = y;
        outb[(size_t)row * DD + c] = (bf16)y;
    }
}

// ---------------------------------------------------------------------------
// Generic bf16 WMMA GEMM:  C[M,N] = A[M,K] (bf16, row major)
//                                   x BT[N,K] (bf16, pre-transposed)
// One wave (32 threads) computes a 32(M) x 64(N) tile: 2 A-frags x 4 B-frags
// -> 8 accumulators (64 VGPRs; fits without spilling, unlike a 64x64 tile).
// Epilogue: + bias, optional GELU, + res1 + res2 (fp32, compact stride N),
// writes fp32 Cf (stride ldc) and/or bf16 Cb (stride ldc).
// grid = (N/64, M/32), block = 32.
// ---------------------------------------------------------------------------
__global__ void gemm_bf16_k(const bf16* __restrict__ A, const bf16* __restrict__ BT,
                            const float* __restrict__ bias,
                            const float* __restrict__ res1, const float* __restrict__ res2,
                            float* __restrict__ Cf, bf16* __restrict__ Cb,
                            int N, int K, int ldc, int do_gelu) {
    int lane = threadIdx.x & 31;
    int half = lane >> 4;
    int l16 = lane & 15;
    int m0 = blockIdx.y * 32;
    int n0 = blockIdx.x * 64;

    const bf16* abase = A + (size_t)(m0 + l16) * K + half * 8;
    const bf16* bbase = BT + (size_t)(n0 + l16) * K + half * 8;

    v8f acc[8];
#pragma unroll
    for (int i = 0; i < 8; i++) acc[i] = v8f{};

    for (int k0 = 0; k0 < K; k0 += 32) {
        v16bf a[2];
#pragma unroll
        for (int mi = 0; mi < 2; mi++) {
            const bf16* ap = abase + (size_t)mi * 16 * K + k0;
            a[mi] = mk16(*(const v8bf*)ap, *(const v8bf*)(ap + 16));
        }
#pragma unroll
        for (int j = 0; j < 4; j++) {
            const bf16* bp = bbase + (size_t)j * 16 * K + k0;
            v16bf b = mk16(*(const v8bf*)bp, *(const v8bf*)(bp + 16));
#pragma unroll
            for (int mi = 0; mi < 2; mi++)
                acc[mi * 4 + j] = wmma_bf16(a[mi], b, acc[mi * 4 + j]);
        }
    }

#pragma unroll
    for (int mi = 0; mi < 2; mi++) {
#pragma unroll
        for (int j = 0; j < 4; j++) {
            int n = n0 + j * 16 + l16;
            float bv = bias ? bias[n] : 0.0f;
#pragma unroll
            for (int r = 0; r < 8; r++) {
                int m = m0 + mi * 16 + r + half * 8;
                float v = acc[mi * 4 + j][r] + bv;
                if (do_gelu) v = gelu_tanh(v);
                size_t ridx = (size_t)m * N + n;
                if (res1) v += res1[ridx];
                if (res2) v += res2[ridx];
                size_t cidx = (size_t)m * ldc + n;
                if (Cf) Cf[cidx] = v;
                if (Cb) Cb[cidx] = (bf16)v;
            }
        }
    }
}

// ---------------------------------------------------------------------------
// Sliding-window attention. grid = (S, H, B), block = 128.
// qkv fp32 [B*S, 3*D]; output bf16 [B*S, D].
// ---------------------------------------------------------------------------
__global__ void swa_k(const float* __restrict__ qkv, bf16* __restrict__ outb) {
    int s = blockIdx.x, h = blockIdx.y, b = blockIdx.z;
    __shared__ float sc[72];
    int t = threadIdx.x;
    size_t rowq = ((size_t)b * SS + s) * (3 * DD);

    if (t < WW + 1) {
        int idx = s + t - WW / 2;
        if (idx < 0 || idx >= SS) {
            sc[t] = -__builtin_inff();
        } else {
            const float* qp = qkv + rowq + h * 64;
            const float* kp = qkv + ((size_t)b * SS + idx) * (3 * DD) + DD + h * 64;
            float acc = 0.0f;
#pragma unroll 8
            for (int d = 0; d < 64; d++) acc += qp[d] * kp[d];
            sc[t] = acc * 0.125f;  // 1/sqrt(64)
        }
    }
    __syncthreads();
    if (t == 0) {
        float mx = -__builtin_inff();
        for (int w = 0; w < WW + 1; w++) mx = fmaxf(mx, sc[w]);
        float ssum = 0.0f;
        for (int w = 0; w < WW + 1; w++) { float e = __expf(sc[w] - mx); sc[w] = e; ssum += e; }
        float inv = 1.0f / ssum;
        for (int w = 0; w < WW + 1; w++) sc[w] *= inv;
    }
    __syncthreads();
    if (t < 64) {
        float acc = 0.0f;
        for (int w = 0; w < WW + 1; w++) {
            int idx = s + w - WW / 2;
            idx = idx < 0 ? 0 : (idx > SS - 1 ? SS - 1 : idx);
            acc += sc[w] * qkv[((size_t)b * SS + idx) * (3 * DD) + 2 * DD + h * 64 + t];
        }
        outb[((size_t)b * SS + s) * DD + h * 64 + t] = (bf16)acc;
    }
}

// ---------------------------------------------------------------------------
// Neural-memory chunked scan. One block per (b, mh): blockIdx.x = b*MH + mh.
// 256 threads (8 waves). Dynamic LDS (192 KB of the 320 KB WGP pool):
//   Mf   [128x128] fp32   (master memory matrix)
//   Momf [128x128] fp32   (momentum)
//   MbfT [128x128] bf16   (M transposed, for B fragments)
//   diffT[128x64]  bf16   ((pred - v) transposed, for B fragments)
//   kT   [128x64]  bf16   (k-chunk transposed, for grad A fragments)
// per chunk: out = q@M, pred = k@M (WMMA), grad = k^T@(pred-v)/CH (WMMA),
// Mom = .9*Mom - .1*grad ; M = .99*M + Mom.
// ---------------------------------------------------------------------------
__global__ void neural_mem_scan_k(const bf16* __restrict__ qb, const bf16* __restrict__ kb,
                                  const bf16* __restrict__ vb, bf16* __restrict__ outb) {
    extern __shared__ char lds[];
    float* Mf    = (float*)lds;                      // 64 KB
    float* Momf  = Mf + 128 * 128;                   // 64 KB
    bf16*  MbfT  = (bf16*)(Momf + 128 * 128);        // 32 KB
    bf16*  diffT = MbfT + 128 * 128;                 // 16 KB
    bf16*  kT    = diffT + 128 * 64;                 // 16 KB

    int b = blockIdx.x >> 2, mh = blockIdx.x & 3;
    int tid = threadIdx.x, wave = tid >> 5, lane = tid & 31;
    int half = lane >> 4, l16 = lane & 15;

    for (int i = tid; i < 128 * 128; i += 256) {
        Mf[i] = 0.0f; Momf[i] = 0.0f; MbfT[i] = (bf16)0.0f;
    }
    __syncthreads();

    const int rs = DD;  // row stride of q/k/v buffers
    for (int n = 0; n < SS / CHH; n++) {
        size_t g0 = (size_t)b * SS + (size_t)n * CHH;

        // ---- stage k chunk transposed: kT[d*64 + c] = k[c][d] ----
        for (int i = tid; i < CHH * 128; i += 256) {
            int c = i >> 7, d = i & 127;
            kT[d * 64 + c] = kb[(g0 + c) * rs + mh * 128 + d];
        }

        // ---- Phase A: out & pred tiles (4 c-tiles x 8 e-tiles = 32 tiles) ----
#pragma unroll
        for (int i = 0; i < 4; i++) {
            int tileId = wave * 4 + i;
            int ct = tileId >> 3, et = tileId & 7;
            int crow = ct * 16 + l16;
            const bf16* qrow = qb + (g0 + crow) * rs + mh * 128 + half * 8;
            const bf16* krow = kb + (g0 + crow) * rs + mh * 128 + half * 8;
            v8f aco = {}, acp = {};
            for (int k0 = 0; k0 < 128; k0 += 32) {
                v16bf aq = mk16(*(const v8bf*)(qrow + k0), *(const v8bf*)(qrow + k0 + 16));
                v16bf ak = mk16(*(const v8bf*)(krow + k0), *(const v8bf*)(krow + k0 + 16));
                const bf16* brow = MbfT + (size_t)(et * 16 + l16) * 128 + k0 + half * 8;
                v16bf bm = mk16(*(const v8bf*)brow, *(const v8bf*)(brow + 16));
                aco = wmma_bf16(aq, bm, aco);
                acp = wmma_bf16(ak, bm, acp);
            }
#pragma unroll
            for (int r = 0; r < 8; r++) {
                int c = ct * 16 + r + half * 8;
                int e = et * 16 + l16;
                size_t gidx = (g0 + c) * rs + mh * 128 + e;
                outb[gidx] = (bf16)aco[r];
                float vv = (float)vb[gidx];
                diffT[(size_t)e * 64 + c] = (bf16)(acp[r] - vv);
            }
        }
        __syncthreads();  // diffT + kT complete

        // ---- Phase B: grad tiles (d-tile = wave, e-tiles 0..7) + update ----
        for (int et = 0; et < 8; et++) {
            int drow = wave * 16 + l16;  // grad row index (d)
            const bf16* arow = kT + (size_t)drow * 64 + half * 8;
            v8f acg = {};
#pragma unroll
            for (int k0 = 0; k0 < 64; k0 += 32) {
                v16bf ak = mk16(*(const v8bf*)(arow + k0), *(const v8bf*)(arow + k0 + 16));
                const bf16* brow = diffT + (size_t)(et * 16 + l16) * 64 + k0 + half * 8;
                v16bf bd = mk16(*(const v8bf*)brow, *(const v8bf*)(brow + 16));
                acg = wmma_bf16(ak, bd, acg);
            }
#pragma unroll
            for (int r = 0; r < 8; r++) {
                int d = wave * 16 + r + half * 8;
                int e = et * 16 + l16;
                int idx = d * 128 + e;
                float g = acg[r] * (1.0f / CHH);
                float mom = 0.9f * Momf[idx] - 0.1f * g;
                Momf[idx] = mom;
                float m = 0.99f * Mf[idx] + mom;
                Mf[idx] = m;
                MbfT[e * 128 + d] = (bf16)m;
            }
        }
        __syncthreads();
    }
}

// ---------------------------------------------------------------------------
// Small helper kernels
// ---------------------------------------------------------------------------
__global__ void pack_bf16_k(const float* __restrict__ in, bf16* __restrict__ out,
                            int n_total, int ncols, int ldo) {
    int i = blockIdx.x * blockDim.x + threadIdx.x;
    if (i >= n_total) return;
    int m = i / ncols, c = i % ncols;
    out[(size_t)m * ldo + c] = (bf16)in[i];
}

__global__ void bcast_bf16_k(const float* __restrict__ vec, bf16* __restrict__ out,
                             int n_total, int ncols, int ldo) {
    int i = blockIdx.x * blockDim.x + threadIdx.x;
    if (i >= n_total) return;
    int m = i / ncols, c = i % ncols;
    out[(size_t)m * ldo + c] = (bf16)vec[c];
}

__global__ void pool_mean_k(const float* __restrict__ pm, float* __restrict__ mean) {
    int j = threadIdx.x;  // 512
    float s = 0.0f;
    for (int t = 0; t < NPTT; t++) s += pm[t * DD + j];
    mean[j] = s * (1.0f / NPTT);
}

__global__ void pooled_proj_k(const float* __restrict__ mean, const float* __restrict__ W,
                              const float* __restrict__ bias, float* __restrict__ pooled) {
    int j = threadIdx.x;  // 512
    float s = bias[j];
    for (int k = 0; k < DD; k++) s += mean[k] * W[k * DD + j];
    pooled[j] = s;
}

// ---------------------------------------------------------------------------
// Host orchestration
// ---------------------------------------------------------------------------
extern "C" void kernel_launch(void* const* d_in, const int* in_sizes, int n_in,
                              void* d_out, int out_size, void* d_ws, size_t ws_size,
                              hipStream_t stream) {
    (void)in_sizes; (void)n_in; (void)out_size; (void)ws_size;

    const float* x        = (const float*)d_in[0];
    const float* n1w      = (const float*)d_in[1];
    const float* n1b      = (const float*)d_in[2];
    const float* qkv_w    = (const float*)d_in[3];
    const float* qkv_b    = (const float*)d_in[4];
    const float* proj_w   = (const float*)d_in[5];
    const float* proj_b   = (const float*)d_in[6];
    const float* n2w      = (const float*)d_in[7];
    const float* n2b      = (const float*)d_in[8];
    const float* fc1_w    = (const float*)d_in[9];
    const float* fc1_b    = (const float*)d_in[10];
    const float* fc2_w    = (const float*)d_in[11];
    const float* fc2_b    = (const float*)d_in[12];
    const float* mln_w    = (const float*)d_in[13];
    const float* mln_b    = (const float*)d_in[14];
    const float* mwq      = (const float*)d_in[15];
    const float* mwk      = (const float*)d_in[16];
    const float* mwv      = (const float*)d_in[17];
    const float* mproj_w  = (const float*)d_in[18];
    const float* mproj_b  = (const float*)d_in[19];
    const float* pm_tok   = (const float*)d_in[20];
    const float* pm_w     = (const float*)d_in[21];
    const float* pm_b     = (const float*)d_in[22];
    const float* fusion_w = (const float*)d_in[23];
    const float* fusion_b = (const float*)d_in[24];

    const int M = BB * SS;  // 4096 rows

    // --- workspace bump allocator ---
    char* wsp = (char*)d_ws;
    auto alloc = [&](size_t bytes) -> char* {
        char* p = wsp;
        wsp += (bytes + 255) & ~(size_t)255;
        return p;
    };
    bf16* wtq  = (bf16*)alloc((size_t)LL * 1536 * 512 * 2);
    bf16* wtp  = (bf16*)alloc((size_t)LL * 512 * 512 * 2);
    bf16* wtf1 = (bf16*)alloc((size_t)LL * 2048 * 512 * 2);
    bf16* wtf2 = (bf16*)alloc((size_t)LL * 512 * 2048 * 2);
    bf16* wtmq = (bf16*)alloc((size_t)512 * 512 * 2);
    bf16* wtmk = (bf16*)alloc((size_t)512 * 512 * 2);
    bf16* wtmv = (bf16*)alloc((size_t)512 * 512 * 2);
    bf16* wtmp = (bf16*)alloc((size_t)512 * 512 * 2);
    bf16* wtfu = (bf16*)alloc((size_t)512 * 1536 * 2);
    float* h0   = (float*)alloc((size_t)M * DD * 4);
    float* h1   = (float*)alloc((size_t)M * DD * 4);
    float* hnf  = (float*)alloc((size_t)M * DD * 4);
    bf16*  lnb  = (bf16*)alloc((size_t)M * DD * 2);
    float* qkvf = (float*)alloc((size_t)M * 3 * DD * 4);
    bf16*  attnb = (bf16*)alloc((size_t)M * DD * 2);
    bf16*  ff1b = (bf16*)alloc((size_t)M * 4 * DD * 2);
    bf16*  nxb  = (bf16*)alloc((size_t)M * DD * 2);
    bf16*  qmb  = (bf16*)alloc((size_t)M * DD * 2);
    bf16*  kmb  = (bf16*)alloc((size_t)M * DD * 2);
    bf16*  vmb  = (bf16*)alloc((size_t)M * DD * 2);
    bf16*  memob = (bf16*)alloc((size_t)M * DD * 2);
    bf16*  fused = (bf16*)alloc((size_t)M * 3 * DD * 2);
    float* meanv  = (float*)alloc(DD * 4);
    float* pooledv = (float*)alloc(DD * 4);

    auto wblocks = [](size_t elems) { return (unsigned)((elems + 255) / 256); };

    // --- 1) weight transposes to bf16 [N,K] ---
    for (int l = 0; l < LL; l++) {
        wtrans_k<<<wblocks((size_t)512 * 1536), 256, 0, stream>>>(
            qkv_w + (size_t)l * 512 * 1536, wtq + (size_t)l * 1536 * 512, 512, 1536);
        wtrans_k<<<wblocks((size_t)512 * 512), 256, 0, stream>>>(
            proj_w + (size_t)l * 512 * 512, wtp + (size_t)l * 512 * 512, 512, 512);
        wtrans_k<<<wblocks((size_t)512 * 2048), 256, 0, stream>>>(
            fc1_w + (size_t)l * 512 * 2048, wtf1 + (size_t)l * 2048 * 512, 512, 2048);
        wtrans_k<<<wblocks((size_t)2048 * 512), 256, 0, stream>>>(
            fc2_w + (size_t)l * 2048 * 512, wtf2 + (size_t)l * 512 * 2048, 2048, 512);
    }
    wtrans_k<<<wblocks((size_t)512 * 512), 256, 0, stream>>>(mwq, wtmq, 512, 512);
    wtrans_k<<<wblocks((size_t)512 * 512), 256, 0, stream>>>(mwk, wtmk, 512, 512);
    wtrans_k<<<wblocks((size_t)512 * 512), 256, 0, stream>>>(mwv, wtmv, 512, 512);
    wtrans_k<<<wblocks((size_t)512 * 512), 256, 0, stream>>>(mproj_w, wtmp, 512, 512);
    wtrans_k<<<wblocks((size_t)1536 * 512), 256, 0, stream>>>(fusion_w, wtfu, 1536, 512);

    const dim3 g512(512 / 64, M / 32);
    const dim3 g1536(1536 / 64, M / 32);
    const dim3 g2048(2048 / 64, M / 32);

    // --- 2) short-term transformer stack ---
    const float* hin = x;
    float* hb[2] = {h0, h1};
    int cur = 0;
    for (int l = 0; l < LL; l++) {
        // attn block
        layernorm_k<<<M, 128, 0, stream>>>(hin, n1w + l * DD, n1b + l * DD, nullptr, lnb);
        gemm_bf16_k<<<g1536, 32, 0, stream>>>(lnb, wtq + (size_t)l * 1536 * 512,
            qkv_b + l * 1536, nullptr, nullptr, qkvf, nullptr, 1536, 512, 1536, 0);
        swa_k<<<dim3(SS, HH, BB), 128, 0, stream>>>(qkvf, attnb);
        float* hmid = hb[cur]; cur ^= 1;
        gemm_bf16_k<<<g512, 32, 0, stream>>>(attnb, wtp + (size_t)l * 512 * 512,
            proj_b + l * DD, hin, nullptr, hmid, nullptr, 512, 512, 512, 0);
        // ffn block: h = hmid + (gelu(hn@fc1+b1)@fc2 + b2 + hn)
        layernorm_k<<<M, 128, 0, stream>>>(hmid, n2w + l * DD, n2b + l * DD, hnf, lnb);
        gemm_bf16_k<<<g2048, 32, 0, stream>>>(lnb, wtf1 + (size_t)l * 2048 * 512,
            fc1_b + l * 2048, nullptr, nullptr, nullptr, ff1b, 2048, 512, 2048, 1);
        float* hnext = hb[cur]; cur ^= 1;
        gemm_bf16_k<<<g512, 32, 0, stream>>>(ff1b, wtf2 + (size_t)l * 512 * 2048,
            fc2_b + l * DD, hnf, hmid, hnext, nullptr, 512, 2048, 512, 0);
        hin = hnext;
    }
    // short_term -> fused cols [0, 512)
    pack_bf16_k<<<wblocks((size_t)M * DD), 256, 0, stream>>>(hin, fused, M * DD, DD, 3 * DD);

    // --- 3) long-term neural memory branch ---
    layernorm_k<<<M, 128, 0, stream>>>(x, mln_w, mln_b, nullptr, nxb);
    gemm_bf16_k<<<g512, 32, 0, stream>>>(nxb, wtmq, nullptr, nullptr, nullptr,
                                         nullptr, qmb, 512, 512, 512, 0);
    gemm_bf16_k<<<g512, 32, 0, stream>>>(nxb, wtmk, nullptr, nullptr, nullptr,
                                         nullptr, kmb, 512, 512, 512, 0);
    gemm_bf16_k<<<g512, 32, 0, stream>>>(nxb, wtmv, nullptr, nullptr, nullptr,
                                         nullptr, vmb, 512, 512, 512, 0);
    // per-(b,head) scan; 192 KB dynamic LDS (CDNA5 WGP has 320 KB)
    size_t scan_lds = (size_t)128 * 128 * 4 * 2   // Mf + Momf
                    + (size_t)128 * 128 * 2       // MbfT
                    + (size_t)128 * 64 * 2        // diffT
                    + (size_t)128 * 64 * 2;       // kT
    neural_mem_scan_k<<<BB * MHH, 256, scan_lds, stream>>>(qmb, kmb, vmb, memob);
    // long_term = x + mem@proj + b -> fused cols [512, 1024)
    gemm_bf16_k<<<g512, 32, 0, stream>>>(memob, wtmp, mproj_b, x, nullptr,
                                         nullptr, fused + 512, 512, 512, 3 * DD, 0);

    // --- 4) persistent branch -> fused cols [1024, 1536) ---
    pool_mean_k<<<1, 512, 0, stream>>>(pm_tok, meanv);
    pooled_proj_k<<<1, 512, 0, stream>>>(meanv, pm_w, pm_b, pooledv);
    bcast_bf16_k<<<wblocks((size_t)M * DD), 256, 0, stream>>>(
        pooledv, fused + 2 * DD, M * DD, DD, 3 * DD);

    // --- 5) fusion GEMM -> d_out (fp32) ---
    gemm_bf16_k<<<g512, 32, 0, stream>>>(fused, wtfu, fusion_b, nullptr, nullptr,
                                         (float*)d_out, nullptr, 512, 1536, 512, 0);
}
